// DiMiMeLayer_44873818308617
// MI455X (gfx1250) — matrix operand order repaired and checked by reading, not compiled, not run on hardware
//
#include <hip/hip_runtime.h>
#include <hip/hip_bf16.h>
#include <math.h>

#define C_NUM 50000
#define B_NUM 30000
#define A_NUM 30000
#define E_BC  600000
#define E_AC  600000
#define E_CC  500000
#define HID   64
#define NEG_SLOPE 0.2f

typedef __attribute__((ext_vector_type(16))) __bf16 v16bf;
typedef __attribute__((ext_vector_type(8)))  __bf16 v8bf;
typedef __attribute__((ext_vector_type(8)))  float  v8f;

// ---------------- helpers ----------------

__device__ __forceinline__ __bf16 f2bf(float f) {
  // round-to-nearest-even f32 -> bf16 (used only in staging kernels, off the hot path)
  unsigned u = __float_as_uint(f);
  unsigned r = u + 0x7FFFu + ((u >> 16) & 1u);
  unsigned short h = (unsigned short)(r >> 16);
  return __builtin_bit_cast(__bf16, h);
}

__device__ __forceinline__ void atomicMaxF(float* addr, float val) {
  // monotonic-in-float-order trick; init value must be -inf
  if (val >= 0.0f) atomicMax((int*)addr, __float_as_int(val));
  else             atomicMin((unsigned int*)addr, (unsigned int)__float_as_int(val));
}

__device__ __forceinline__ float leaky(float x) {
  return x > 0.0f ? x : NEG_SLOPE * x;
}

// ---------------- staging: f32 -> bf16 row-major copy ----------------

__global__ __launch_bounds__(256)
void cvt_f32_to_bf16(const float* __restrict__ src, __bf16* __restrict__ dst, int n)
{
  int i = (blockIdx.x * 256 + threadIdx.x) * 8;
  if (i >= n) return;
  v8bf o;
#pragma unroll
  for (int j = 0; j < 8; ++j) o[j] = f2bf(src[i + j]);
  *(v8bf*)(dst + i) = o;
}

// ---------------- staging: pack W[K,M] into WMMA B-fragment image ----------------
// layout: Wp[kt][tm][lane][j], j=0..15; lane&15 = column-in-tile, lane>>4 picks K half:
// element j of lane holds W[kt*32 + (lane>>4)*16 + j][tm*16 + (lane&15)]  (contiguous k order, same as the A fragment)

__global__ __launch_bounds__(256)
void pack_w_bf16(const float* __restrict__ W, __bf16* __restrict__ Wp, int K, int M)
{
  int idx = blockIdx.x * 256 + threadIdx.x;
  if (idx >= K * M) return;
  int j    = idx & 15;
  int lane = (idx >> 4) & 31;
  int tm   = (idx >> 9) % (M >> 4);
  int kt   = idx / ((M >> 4) << 9);
  int k    = kt * 32 + (lane >> 4) * 16 + j;
  int col  = tm * 16 + (lane & 15);
  Wp[idx] = f2bf(W[(size_t)k * M + col]);
}

// ---------------- dense GEMM: Y[N,M] = Xbf[N,K] @ W (+bias), bf16 WMMA ----------------
// one wave per 16x32 output tile (two independent 16x16 accumulator chains sharing the
// A fragment -> second WMMA chain fills the bf16 WMMA->WMMA hazard slots); fully
// unrolled over K; pure load/load/load/wmma/wmma inner loop

template<int K>
__global__ __launch_bounds__(256)
void gemm_bf16_wmma(const __bf16* __restrict__ Xbf, const __bf16* __restrict__ Wp,
                    const float* __restrict__ bias, float* __restrict__ Y,
                    int N, int M, int ldY, int colOff)
{
  int wave    = (blockIdx.x * 256 + threadIdx.x) >> 5;
  int tilesM  = M >> 4;           // 16-wide column tiles
  int pairsM  = M >> 5;           // processed two at a time
  int tileN   = wave / pairsM;
  int tm0     = (wave - tileN * pairsM) * 2;
  if (tileN * 16 >= N) return;    // whole-wave uniform: EXEC stays all-ones
  int lane = threadIdx.x & 31;
  int r    = lane & 15;
  int hi   = lane >> 4;

  const __bf16* Xrow = Xbf + (size_t)(tileN * 16 + r) * K;
  const __bf16* Bp0  = Wp + ((size_t)tm0 * 32 + lane) * 16;
  const __bf16* Bp1  = Bp0 + 512;                 // adjacent column tile
  const size_t  Bstep = (size_t)tilesM * 512;     // one kt step in panel

  v8f acc0 = {};
  v8f acc1 = {};
#pragma unroll
  for (int kt = 0; kt < K / 32; ++kt) {
    // A fragment: lanes0-15 K=k0..15 ; lanes16-31 K=k0+16..31
    v8bf alo = *(const v8bf*)(Xrow + kt * 32 + hi * 16);
    v8bf ahi = *(const v8bf*)(Xrow + kt * 32 + hi * 16 + 8);
    v16bf a = __builtin_shufflevector(alo, ahi, 0,1,2,3,4,5,6,7,8,9,10,11,12,13,14,15);
    v16bf b0 = *(const v16bf*)(Bp0 + kt * Bstep);
    v16bf b1 = *(const v16bf*)(Bp1 + kt * Bstep);
    acc0 = __builtin_amdgcn_wmma_f32_16x16x32_bf16(false, a, false, b0,
                                                   (short)0, acc0, false, false);
    acc1 = __builtin_amdgcn_wmma_f32_16x16x32_bf16(false, a, false, b1,
                                                   (short)0, acc1, false, false);
  }

  float bv0 = bias ? bias[tm0 * 16 + r] : 0.0f;
  float bv1 = bias ? bias[tm0 * 16 + 16 + r] : 0.0f;
  float* Yt = Y + (size_t)(tileN * 16 + hi * 8) * ldY + colOff + tm0 * 16 + r;
#pragma unroll
  for (int q = 0; q < 8; ++q) {          // D VGPR q -> row (q + 8*hi), col (lane&15)
    Yt[(size_t)q * ldY]      = acc0[q] + bv0;
    Yt[(size_t)q * ldY + 16] = acc1[q] + bv1;
  }
}

// ---------------- per-node attention scalars: out[i,h] = dot(H[i,h,:], att[h,:]) ----------------

__global__ __launch_bounds__(256)
void node_dot(const float* __restrict__ H, const float* __restrict__ att,
              float* __restrict__ out, int n, int heads, int do_leaky)
{
  int wid  = (blockIdx.x * 256 + threadIdx.x) >> 5;
  int lane = threadIdx.x & 31;
  if (wid >= n * heads) return;
  int node = wid / heads;
  int h    = wid - node * heads;
  float2 hv = ((const float2*)(H + (size_t)node * heads * HID + h * HID))[lane];
  float2 av = ((const float2*)(att + h * HID))[lane];
  float s = hv.x * av.x + hv.y * av.y;
#pragma unroll
  for (int off = 16; off > 0; off >>= 1) s += __shfl_xor(s, off, 32);
  if (lane == 0) {
    if (do_leaky) s = leaky(s);
    out[wid] = s;
  }
}

__global__ void fill_f32(float* __restrict__ p, float v, int n) {
  int i = blockIdx.x * blockDim.x + threadIdx.x;
  if (i < n) p[i] = v;
}

// ---------------- bipartite GAT edge passes (heads==1, alpha = s[srcn]) ----------------

__global__ void bi_seg_max(const int* __restrict__ srcn, const int* __restrict__ seg,
                           const float* __restrict__ s, float* __restrict__ amax, int E)
{
  int e = blockIdx.x * blockDim.x + threadIdx.x;
  if (e < E) atomicMaxF(&amax[seg[e]], s[srcn[e]]);
}

__global__ void bi_seg_sum(const int* __restrict__ srcn, const int* __restrict__ seg,
                           const float* __restrict__ s, const float* __restrict__ amax,
                           float* __restrict__ ssum, int E)
{
  int e = blockIdx.x * blockDim.x + threadIdx.x;
  if (e < E) {
    int t = seg[e];
    atomicAdd(&ssum[t], __expf(s[srcn[e]] - amax[t]));
  }
}

__global__ __launch_bounds__(256)
void bi_scatter(const int* __restrict__ srcn, const int* __restrict__ seg,
                const float* __restrict__ s, const float* __restrict__ amax,
                const float* __restrict__ ssum, const float* __restrict__ Hmsg,
                float* __restrict__ aggr, int E)
{
  int e = (blockIdx.x * 256 + threadIdx.x) >> 5;
  if (e >= E) return;
  int lane = threadIdx.x & 31;
  int sn = srcn[e], t = seg[e];
  float a = __expf(s[sn] - amax[t]) / (ssum[t] + 1e-16f);
  float2 m = ((const float2*)(Hmsg + (size_t)sn * HID))[lane];
  float* o = aggr + (size_t)t * HID + lane * 2;
  atomicAdd(o,     m.x * a);
  atomicAdd(o + 1, m.y * a);
}

__global__ void bi_final(const float* __restrict__ aggr, const float* __restrict__ bias,
                         const float* __restrict__ resid, float* __restrict__ out,
                         int n, int ldOut, int colOff)
{
  int i = blockIdx.x * blockDim.x + threadIdx.x;
  if (i >= n * HID) return;
  int node = i >> 6, f = i & 63;
  float v = aggr[i] + bias[f] + resid[i];
  out[(size_t)node * ldOut + colOff + f] = v > 0.0f ? v : 0.0f;   // relu
}

// ---------------- homogeneous GATConv passes (self-loops appended as virtual edges) ----------------

__global__ void gat_seg_max(const int* __restrict__ e0, const int* __restrict__ e1,
                            const float* __restrict__ asrc, const float* __restrict__ adst,
                            float* __restrict__ amax, int E, int n, int heads)
{
  int ve = blockIdx.x * blockDim.x + threadIdx.x;
  if (ve >= E + n) return;
  int s = ve < E ? e0[ve] : ve - E;
  int t = ve < E ? e1[ve] : ve - E;
  for (int h = 0; h < heads; ++h) {
    float al = leaky(asrc[s * heads + h] + adst[t * heads + h]);
    atomicMaxF(&amax[t * heads + h], al);
  }
}

__global__ void gat_seg_sum(const int* __restrict__ e0, const int* __restrict__ e1,
                            const float* __restrict__ asrc, const float* __restrict__ adst,
                            const float* __restrict__ amax, float* __restrict__ ssum,
                            int E, int n, int heads)
{
  int ve = blockIdx.x * blockDim.x + threadIdx.x;
  if (ve >= E + n) return;
  int s = ve < E ? e0[ve] : ve - E;
  int t = ve < E ? e1[ve] : ve - E;
  for (int h = 0; h < heads; ++h) {
    float al = leaky(asrc[s * heads + h] + adst[t * heads + h]);
    atomicAdd(&ssum[t * heads + h], __expf(al - amax[t * heads + h]));
  }
}

__global__ __launch_bounds__(256)
void gat_scatter(const int* __restrict__ e0, const int* __restrict__ e1,
                 const float* __restrict__ asrc, const float* __restrict__ adst,
                 const float* __restrict__ amax, const float* __restrict__ ssum,
                 const float* __restrict__ Hmsg, float* __restrict__ acc,
                 int E, int n, int heads, float invHeads)
{
  int ve = (blockIdx.x * 256 + threadIdx.x) >> 5;
  if (ve >= E + n) return;
  int lane = threadIdx.x & 31;
  int s = ve < E ? e0[ve] : ve - E;
  int t = ve < E ? e1[ve] : ve - E;
  const float2* mp = (const float2*)(Hmsg + (size_t)s * heads * HID);
  float* op = acc + (size_t)t * HID + lane * 2;
  for (int h = 0; h < heads; ++h) {
    float al = leaky(asrc[s * heads + h] + adst[t * heads + h]);
    float a  = __expf(al - amax[t * heads + h]) / (ssum[t * heads + h] + 1e-16f) * invHeads;
    float2 m = mp[h * 32 + lane];
    atomicAdd(op,     m.x * a);   // head-mean folded into invHeads
    atomicAdd(op + 1, m.y * a);
  }
}

__global__ void gat_final(const float* __restrict__ acc, const float* __restrict__ bias,
                          float* __restrict__ out, int n, int act)
{
  int i = blockIdx.x * blockDim.x + threadIdx.x;
  if (i >= n * HID) return;
  int f = i & 63;
  float v = acc[i] + bias[f];
  out[i] = (act == 1) ? (v > 0.0f ? v : 0.0f) : tanhf(v);
}

// ---------------- host orchestration ----------------

static inline int cdiv(int a, int b) { return (a + b - 1) / b; }

extern "C" void kernel_launch(void* const* d_in, const int* in_sizes, int n_in,
                              void* d_out, int out_size, void* d_ws, size_t ws_size,
                              hipStream_t stream)
{
  (void)in_sizes; (void)n_in; (void)out_size; (void)ws_size;

  const int* CC0 = (const int*)d_in[0]; const int* CC1 = CC0 + E_CC;
  const int* BC0 = (const int*)d_in[1]; const int* BC1 = BC0 + E_BC;
  const int* AC0 = (const int*)d_in[2]; const int* AC1 = AC0 + E_AC;
  const float* Bf = (const float*)d_in[3];
  const float* Af = (const float*)d_in[4];
  const float* Cf = (const float*)d_in[5];
  const float *Wi_BC=(const float*)d_in[6],  *bi_BC=(const float*)d_in[7],
              *Wj_BC=(const float*)d_in[8],  *bj_BC=(const float*)d_in[9],
              *att_BC=(const float*)d_in[10],*bias_BC=(const float*)d_in[11];
  const float *Wi_AC=(const float*)d_in[12], *bi_AC=(const float*)d_in[13],
              *Wj_AC=(const float*)d_in[14], *bj_AC=(const float*)d_in[15],
              *att_AC=(const float*)d_in[16],*bias_AC=(const float*)d_in[17];
  const float *W1=(const float*)d_in[18], *att1_src=(const float*)d_in[19],
              *att1_dst=(const float*)d_in[20], *bias1=(const float*)d_in[21];
  const float *W2=(const float*)d_in[22], *att2_src=(const float*)d_in[23],
              *att2_dst=(const float*)d_in[24], *bias2=(const float*)d_in[25];
  const float *Wi_CB=(const float*)d_in[26], *bi_CB=(const float*)d_in[27],
              *Wj_CB=(const float*)d_in[28], *bj_CB=(const float*)d_in[29],
              *att_CB=(const float*)d_in[30],*bias_CB=(const float*)d_in[31];
  const float *Wi_CA=(const float*)d_in[32], *bi_CA=(const float*)d_in[33],
              *Wj_CA=(const float*)d_in[34], *bj_CA=(const float*)d_in[35],
              *att_CA=(const float*)d_in[36],*bias_CA=(const float*)d_in[37];

  float* w    = (float*)d_ws;
  float* Ccat = w;                                   // [50000,128]      region0 (6.4M f32)
  float* R1   = w + (size_t)6400000;                 // big scratch      (12.8M f32)
  float* R2   = w + (size_t)19200000;                // scalar scratch   (1.0M f32)
  __bf16* XBF = (__bf16*)(w + (size_t)20200000);     // bf16 X staging   (3.2M f32 slots)
  __bf16* WP  = (__bf16*)(w + (size_t)23400000);     // bf16 W panel     (16K f32 slots)

  float* out  = (float*)d_out;
  float* Bout = out;
  float* Aout = out + (size_t)B_NUM * HID;
  float* Cout = out + (size_t)(B_NUM + A_NUM) * HID; // final C embedding

  auto gemm = [&](const float* X, const float* Wm, const float* b, float* Y,
                  int N, int K, int M, int ldY, int colOff) {
    cvt_f32_to_bf16<<<cdiv(N * K / 8, 256), 256, 0, stream>>>(X, XBF, N * K);
    pack_w_bf16<<<cdiv(K * M, 256), 256, 0, stream>>>(Wm, WP, K, M);
    int blocks = cdiv((N / 16) * (M / 32) * 32, 256);
    if (K == 128)
      gemm_bf16_wmma<128><<<blocks, 256, 0, stream>>>(XBF, WP, b, Y, N, M, ldY, colOff);
    else
      gemm_bf16_wmma<64><<<blocks, 256, 0, stream>>>(XBF, WP, b, Y, N, M, ldY, colOff);
  };

  // ---- stage 1: B->C and A->C bipartite GAT, relu, concat into Ccat ----
  auto stage1 = [&](const float* featS, int nS, const float* Wi, const float* bi,
                    const float* Wj, const float* bj, const float* att, const float* bias,
                    const int* eS, const int* eD, int E, int colOff) {
    float* Hs   = R1;                               // [nS,64]
    float* Hd   = R1 + (size_t)B_NUM * HID;         // [50000,64]
    float* aggr = Hd + (size_t)C_NUM * HID;         // [50000,64]
    float* sS   = R2;                               // [nS]
    float* amax = R2 + B_NUM;                       // [50000]
    float* ssum = amax + C_NUM;                     // [50000]
    gemm(featS, Wi, bi, Hs, nS, 128, HID, HID, 0);
    gemm(Cf,    Wj, bj, Hd, C_NUM, 128, HID, HID, 0);
    node_dot<<<cdiv(nS * 32, 256), 256, 0, stream>>>(Hs, att, sS, nS, 1, 1);
    fill_f32<<<cdiv(C_NUM, 256), 256, 0, stream>>>(amax, -INFINITY, C_NUM);
    fill_f32<<<cdiv(C_NUM, 256), 256, 0, stream>>>(ssum, 0.0f, C_NUM);
    fill_f32<<<cdiv(C_NUM * HID, 256), 256, 0, stream>>>(aggr, 0.0f, C_NUM * HID);
    bi_seg_max<<<cdiv(E, 256), 256, 0, stream>>>(eS, eD, sS, amax, E);
    bi_seg_sum<<<cdiv(E, 256), 256, 0, stream>>>(eS, eD, sS, amax, ssum, E);
    bi_scatter<<<cdiv(E * 32, 256), 256, 0, stream>>>(eS, eD, sS, amax, ssum, Hs, aggr, E);
    bi_final<<<cdiv(C_NUM * HID, 256), 256, 0, stream>>>(aggr, bias, Hd, Ccat, C_NUM, 128, colOff);
  };
  stage1(Bf, B_NUM, Wi_BC, bi_BC, Wj_BC, bj_BC, att_BC, bias_BC, BC0, BC1, E_BC, 0);
  stage1(Af, A_NUM, Wi_AC, bi_AC, Wj_AC, bj_AC, att_AC, bias_AC, AC0, AC1, E_AC, HID);

  // ---- stage 2a: gat_conv heads=4 over C_C (+self loops), relu ----
  {
    float* H1   = R1;                               // [50000,256]
    float* asrc = R2;                               // [50000,4]
    float* adst = asrc + C_NUM * 4;
    float* amax = adst + C_NUM * 4;
    float* ssum = amax + C_NUM * 4;
    float* C2   = Ccat;                             // reuse region0 (Ccat dead after GEMM)
    int VE = E_CC + C_NUM;
    gemm(Ccat, W1, nullptr, H1, C_NUM, 128, 256, 256, 0);
    node_dot<<<cdiv(C_NUM * 4 * 32, 256), 256, 0, stream>>>(H1, att1_src, asrc, C_NUM, 4, 0);
    node_dot<<<cdiv(C_NUM * 4 * 32, 256), 256, 0, stream>>>(H1, att1_dst, adst, C_NUM, 4, 0);
    fill_f32<<<cdiv(C_NUM * 4, 256), 256, 0, stream>>>(amax, -INFINITY, C_NUM * 4);
    fill_f32<<<cdiv(C_NUM * 4, 256), 256, 0, stream>>>(ssum, 0.0f, C_NUM * 4);
    fill_f32<<<cdiv(C_NUM * HID, 256), 256, 0, stream>>>(C2, 0.0f, C_NUM * HID);
    gat_seg_max<<<cdiv(VE, 256), 256, 0, stream>>>(CC0, CC1, asrc, adst, amax, E_CC, C_NUM, 4);
    gat_seg_sum<<<cdiv(VE, 256), 256, 0, stream>>>(CC0, CC1, asrc, adst, amax, ssum, E_CC, C_NUM, 4);
    gat_scatter<<<cdiv(VE * 32, 256), 256, 0, stream>>>(CC0, CC1, asrc, adst, amax, ssum,
                                                        H1, C2, E_CC, C_NUM, 4, 0.25f);
    gat_final<<<cdiv(C_NUM * HID, 256), 256, 0, stream>>>(C2, bias1, C2, C_NUM, 1); // relu in place
  }

  // ---- stage 2b: gat_conv heads=1 over C_C (+self loops), tanh -> Cout ----
  {
    float* C2   = Ccat;
    float* H2   = R1;                               // [50000,64]
    float* Cacc = Ccat + (size_t)C_NUM * HID;       // second half of region0
    float* asrc = R2;
    float* adst = asrc + C_NUM;
    float* amax = adst + C_NUM;
    float* ssum = amax + C_NUM;
    int VE = E_CC + C_NUM;
    gemm(C2, W2, nullptr, H2, C_NUM, 64, 64, 64, 0);
    node_dot<<<cdiv(C_NUM * 32, 256), 256, 0, stream>>>(H2, att2_src, asrc, C_NUM, 1, 0);
    node_dot<<<cdiv(C_NUM * 32, 256), 256, 0, stream>>>(H2, att2_dst, adst, C_NUM, 1, 0);
    fill_f32<<<cdiv(C_NUM, 256), 256, 0, stream>>>(amax, -INFINITY, C_NUM);
    fill_f32<<<cdiv(C_NUM, 256), 256, 0, stream>>>(ssum, 0.0f, C_NUM);
    fill_f32<<<cdiv(C_NUM * HID, 256), 256, 0, stream>>>(Cacc, 0.0f, C_NUM * HID);
    gat_seg_max<<<cdiv(VE, 256), 256, 0, stream>>>(CC0, CC1, asrc, adst, amax, E_CC, C_NUM, 1);
    gat_seg_sum<<<cdiv(VE, 256), 256, 0, stream>>>(CC0, CC1, asrc, adst, amax, ssum, E_CC, C_NUM, 1);
    gat_scatter<<<cdiv(VE * 32, 256), 256, 0, stream>>>(CC0, CC1, asrc, adst, amax, ssum,
                                                        H2, Cacc, E_CC, C_NUM, 1, 1.0f);
    gat_final<<<cdiv(C_NUM * HID, 256), 256, 0, stream>>>(Cacc, bias2, Cout, C_NUM, 2); // tanh
  }

  // ---- stage 3: C->B and C->A bipartite GAT (target_to_source), relu ----
  auto stage3 = [&](const float* featS, int nT, const float* Wi, const float* bi,
                    const float* Wj, const float* bj, const float* att, const float* bias,
                    const int* eT, const int* eC, int E, float* outP) {
    float* Hi   = R1;                               // [nT,64]  (resid)
    float* Hj   = R1 + (size_t)B_NUM * HID;         // [50000,64] (messages, from C)
    float* aggr = Hj + (size_t)C_NUM * HID;         // [nT,64]
    float* sC   = R2;                               // [50000]
    float* amax = R2 + C_NUM;                       // [nT]
    float* ssum = amax + B_NUM;                     // [nT]
    gemm(featS, Wi, bi, Hi, nT, 128, HID, HID, 0);
    gemm(Cout,  Wj, bj, Hj, C_NUM, HID, HID, HID, 0);
    node_dot<<<cdiv(C_NUM * 32, 256), 256, 0, stream>>>(Hj, att, sC, C_NUM, 1, 1);
    fill_f32<<<cdiv(nT, 256), 256, 0, stream>>>(amax, -INFINITY, nT);
    fill_f32<<<cdiv(nT, 256), 256, 0, stream>>>(ssum, 0.0f, nT);
    fill_f32<<<cdiv(nT * HID, 256), 256, 0, stream>>>(aggr, 0.0f, nT * HID);
    bi_seg_max<<<cdiv(E, 256), 256, 0, stream>>>(eC, eT, sC, amax, E);
    bi_seg_sum<<<cdiv(E, 256), 256, 0, stream>>>(eC, eT, sC, amax, ssum, E);
    bi_scatter<<<cdiv(E * 32, 256), 256, 0, stream>>>(eC, eT, sC, amax, ssum, Hj, aggr, E);
    bi_final<<<cdiv(nT * HID, 256), 256, 0, stream>>>(aggr, bias, Hi, outP, nT, HID, 0);
  };
  stage3(Bf, B_NUM, Wi_CB, bi_CB, Wj_CB, bj_CB, att_CB, bias_CB, BC0, BC1, E_BC, Bout);
  stage3(Af, A_NUM, Wi_CA, bi_CA, Wj_CA, bj_CA, att_CA, bias_CA, AC0, AC1, E_AC, Aout);
}
